// D2GNN_86964497809683
// MI455X (gfx1250) — compile-verified, matrix operand order checked by reading, not run on hardware
//
#include <hip/hip_runtime.h>
#include <math.h>
#include <stdint.h>

typedef _Float16 v16h __attribute__((ext_vector_type(16)));
typedef _Float16 v8h  __attribute__((ext_vector_type(8)));
typedef float    v8f  __attribute__((ext_vector_type(8)));

constexpr int NN = 65536;        // nodes
constexpr int HD = 128;          // hidden
constexpr int EE = 1048576;      // edges
constexpr int CC = 6;            // classes
constexpr int MM = 2 * NN;       // graph nodes

// ---------------------------------------------------------------------------
// Weight pre-conversion: W[K][128] f32 -> Wt[128][Kp] f16 (transposed,
// K padded to Kp = ceil32(K) with zeros). Makes every GEMM B-tile full,
// 16B-aligned, and f16 so it can be copied global->LDS with async b128 ops.
// ---------------------------------------------------------------------------
__global__ void cvt_wt(const float* __restrict__ W, _Float16* __restrict__ Wt,
                       int K, int Kp)
{
    long i = (long)blockIdx.x * blockDim.x + threadIdx.x;
    if (i >= (long)Kp * 128) return;
    int c = (int)(i & 127), k = (int)(i >> 7);
    Wt[(long)c * Kp + k] = (k < K) ? (_Float16)W[(long)k * 128 + c] : (_Float16)0.0f;
}

// fragment = two aligned 16B LDS loads (ds_load_b128), per ISA 16-bit layout:
// halves 0..7 -> K = hh*8 + 0..7 ; halves 8..15 -> K = 16 + hh*8 + 0..7
static __device__ __forceinline__ v16h frag_ld(const _Float16* row, int hh)
{
    v8h lo = *(const v8h*)(row + hh * 8);
    v8h hi = *(const v8h*)(row + 16 + hh * 8);
    return __builtin_shufflevector(lo, hi, 0, 1, 2, 3, 4, 5, 6, 7,
                                   8, 9, 10, 11, 12, 13, 14, 15);
}

// ---------------------------------------------------------------------------
// GEMM: C[M x 128] = act(A[M x K] @ W[K x 128] + bias)
// A f32 row-major (stride lda); Wt f16 transposed [128][Kp]. 256 thr = 8 waves
// per 128x128 tile; 8 v_wmma_f32_16x16x32_f16 per K-step per wave.
// B tile staged with global_load_async_to_lds_b128 (ASYNCcnt).
// ---------------------------------------------------------------------------
__global__ __launch_bounds__(256)
void gemm_h128(const float* __restrict__ A, int lda, int K,
               const _Float16* __restrict__ Wt, const float* __restrict__ bias,
               float* __restrict__ C, int act)
{
    __shared__ __align__(16) _Float16 sA[128][40];    // 128 rows x 32 k (pad 80B)
    __shared__ __align__(16) _Float16 sBt[128][40];   // 128 cols x 32 k (pad 80B)

    const int tid  = threadIdx.x;
    const int wave = tid >> 5;
    const int lane = tid & 31;
    const int hh   = lane >> 4;
    const int l16  = lane & 15;
    const long rowBase = (long)blockIdx.x * 128;
    const int Kp = (K + 31) & ~31;

    v8f acc[8];
#pragma unroll
    for (int t = 0; t < 8; ++t)
#pragma unroll
        for (int i = 0; i < 8; ++i) acc[t][i] = 0.0f;

    for (int k0 = 0; k0 < Kp; k0 += 32) {
        if (k0 + 32 <= K) {
            // full step: vectorized f32 loads, packed f16 b128 LDS stores
            for (int i = tid; i < 512; i += 256) {
                int r = i >> 2, c8 = (i & 3) * 8;
                const float* src = A + (rowBase + r) * (long)lda + k0 + c8;
                float4 a0 = *(const float4*)src;
                float4 a1 = *(const float4*)(src + 4);
                v8h o;
                o[0] = (_Float16)a0.x; o[1] = (_Float16)a0.y;
                o[2] = (_Float16)a0.z; o[3] = (_Float16)a0.w;
                o[4] = (_Float16)a1.x; o[5] = (_Float16)a1.y;
                o[6] = (_Float16)a1.z; o[7] = (_Float16)a1.w;
                *(v8h*)&sA[r][c8] = o;
            }
        } else {
            // K tail (only the K=100 encoder hits this): checked scalar path
            for (int i = tid; i < 128 * 32; i += 256) {
                int r = i >> 5, c = i & 31, gk = k0 + c;
                sA[r][c] = (gk < K) ? (_Float16)A[(rowBase + r) * (long)lda + gk]
                                    : (_Float16)0.0f;
            }
        }
        // B tile: 128 cols x 64B of pre-transposed f16 weights, async -> LDS
        for (int i = tid; i < 512; i += 256) {
            int c = i >> 2, p = (i & 3) * 8;
            unsigned lo = (unsigned)(uintptr_t)&sBt[c][p];
            unsigned long long ga =
                (unsigned long long)(uintptr_t)(Wt + (long)c * Kp + k0 + p);
            asm volatile("global_load_async_to_lds_b128 %0, %1, off"
                         :: "v"(lo), "v"(ga) : "memory");
        }
        asm volatile("s_wait_asynccnt 0x0" ::: "memory");
        __syncthreads();

        v16h afrag = frag_ld(&sA[wave * 16 + l16][0], hh);
#pragma unroll
        for (int t = 0; t < 8; ++t) {
            v16h bfrag = frag_ld(&sBt[t * 16 + l16][0], hh);
            acc[t] = __builtin_amdgcn_wmma_f32_16x16x32_f16(
                false, afrag, false, bfrag, (short)0, acc[t], false, false);
        }
        __syncthreads();
    }

    // epilogue: C/D layout -> lane gives col, VGPR v gives row v + hh*8
#pragma unroll
    for (int t = 0; t < 8; ++t) {
        int col = t * 16 + l16;
        float b = bias ? bias[col] : 0.0f;
#pragma unroll
        for (int v = 0; v < 8; ++v) {
            long row = rowBase + wave * 16 + hh * 8 + v;
            float x = acc[t][v] + b;
            if (act) x = fmaxf(x, 0.0f);
            C[row * HD + col] = x;
        }
    }
}

// ---------------------------------------------------------------------------
// Elementwise / utility kernels
// ---------------------------------------------------------------------------
__global__ void fill_f32(float* p, float val, long n) {
    long i = (long)blockIdx.x * blockDim.x + threadIdx.x;
    if (i < n) p[i] = val;
}
__global__ void copy_f32(const float* __restrict__ s, float* __restrict__ d, long n) {
    long i = (long)blockIdx.x * blockDim.x + threadIdx.x;
    if (i < n) d[i] = s[i];
}
__global__ void relu_f32(float* p, long n) {
    long i = (long)blockIdx.x * blockDim.x + threadIdx.x;
    if (i < n) p[i] = fmaxf(p[i], 0.0f);
}

static __device__ __forceinline__ float4 ld4(const float* p, int lane) {
    return ((const float4*)p)[lane];
}
static __device__ __forceinline__ float dot4(float4 a, float4 b) {
    return a.x * b.x + a.y * b.y + a.z * b.z + a.w * b.w;
}
static __device__ __forceinline__ void atomicMaxF(float* addr, float val) {
    int* ia = reinterpret_cast<int*>(addr);
    int old = __float_as_int(*addr);
    while (__int_as_float(old) < val) {
        int assumed = old;
        old = atomicCAS(ia, assumed, __float_as_int(val));
        if (old == assumed) break;
    }
}

// ---------------------------------------------------------------------------
// Gating fusion: one wave per node
// ---------------------------------------------------------------------------
__global__ __launch_bounds__(256)
void gating_kernel(const float* __restrict__ jc, const float* __restrict__ ja,
                   const float* __restrict__ jl, const float* __restrict__ jv,
                   const float* __restrict__ dwc, const float* __restrict__ dwa,
                   const float* __restrict__ dwl, const float* __restrict__ dwv,
                   const float* __restrict__ waW, const float* __restrict__ wab,
                   const float* __restrict__ wlW, const float* __restrict__ wlb,
                   const float* __restrict__ wvW, const float* __restrict__ wvb,
                   float* __restrict__ jf)
{
    int gw = (int)(((long)blockIdx.x * blockDim.x + threadIdx.x) >> 5);
    int lane = threadIdx.x & 31;
    if (gw >= NN) return;
    long ro = (long)gw * HD;

    float4 c = ld4(dwc + ro, lane), a = ld4(dwa + ro, lane);
    float4 l = ld4(dwl + ro, lane), v = ld4(dwv + ro, lane);
    float sa = dot4(c, ld4(waW, lane)) + dot4(a, ld4(waW + 128, lane));
    float sl = dot4(c, ld4(wlW, lane)) + dot4(l, ld4(wlW + 128, lane));
    float sv = dot4(c, ld4(wvW, lane)) + dot4(v, ld4(wvW + 128, lane));
#pragma unroll
    for (int m = 16; m >= 1; m >>= 1) {
        sa += __shfl_xor(sa, m, 32);
        sl += __shfl_xor(sl, m, 32);
        sv += __shfl_xor(sv, m, 32);
    }
    float ta = tanhf(sa + wab[0]), tl = tanhf(sl + wlb[0]), tv = tanhf(sv + wvb[0]);
    float mx = fmaxf(ta, fmaxf(tl, tv));
    float ea = expf(ta - mx), el = expf(tl - mx), ev = expf(tv - mx);
    float s = ea + el + ev;
    float w0 = ea / s, w1 = el / s, w2 = ev / s;

    float4 jc4 = ld4(jc + ro, lane), ja4 = ld4(ja + ro, lane);
    float4 jl4 = ld4(jl + ro, lane), jv4 = ld4(jv + ro, lane);
    float4 o;
    o.x = jc4.x + w0 * ja4.x + w1 * jl4.x + w2 * jv4.x;
    o.y = jc4.y + w0 * ja4.y + w1 * jl4.y + w2 * jv4.y;
    o.z = jc4.z + w0 * ja4.z + w1 * jl4.z + w2 * jv4.z;
    o.w = jc4.w + w0 * ja4.w + w1 * jl4.w + w2 * jv4.w;
    ((float4*)(jf + ro))[lane] = o;
}

// ---------------------------------------------------------------------------
// RGCN edge kernels
// ---------------------------------------------------------------------------
__global__ void deg_kernel(const int* __restrict__ dst, float* __restrict__ deg) {
    long e = (long)blockIdx.x * blockDim.x + threadIdx.x;
    if (e < EE) atomicAdd(&deg[dst[e]], 1.0f);
}

__global__ __launch_bounds__(256)
void rgcn_scatter(const float* __restrict__ tmp, float* __restrict__ agg,
                  const int* __restrict__ src, const int* __restrict__ dst,
                  const int* __restrict__ typ, int r)
{
    long idx = (long)blockIdx.x * blockDim.x + threadIdx.x;
    int e = (int)(idx >> 5), q = (int)(idx & 31);
    if (e >= EE || typ[e] != r) return;
    float4 v = ((const float4*)(tmp + (long)src[e] * HD))[q];
    float* o = agg + (long)dst[e] * HD + q * 4;
    atomicAdd(o + 0, v.x); atomicAdd(o + 1, v.y);
    atomicAdd(o + 2, v.z); atomicAdd(o + 3, v.w);
}

__global__ void rgcn_combine(const float* __restrict__ agg, const float* __restrict__ self,
                             const float* __restrict__ deg, float* __restrict__ out,
                             int relu, long n)
{
    long i = (long)blockIdx.x * blockDim.x + threadIdx.x;
    if (i >= n) return;
    long row = i >> 7;
    float x = agg[i] / fmaxf(deg[row], 1.0f) + self[i];
    if (relu) x = fmaxf(x, 0.0f);
    out[i] = x;
}

// ---------------------------------------------------------------------------
// GAT kernels
// ---------------------------------------------------------------------------
__global__ __launch_bounds__(256)
void gat_scores(const float* __restrict__ h, const float* __restrict__ as,
                const float* __restrict__ ad, float* __restrict__ hs,
                float* __restrict__ hd, int mrows)
{
    int gw = (int)(((long)blockIdx.x * blockDim.x + threadIdx.x) >> 5);
    int lane = threadIdx.x & 31;
    if (gw >= mrows) return;
    float4 hv = ld4(h + (long)gw * HD, lane);
    float ps = dot4(hv, ld4(as, lane));
    float pd = dot4(hv, ld4(ad, lane));
#pragma unroll
    for (int m = 16; m >= 1; m >>= 1) {
        ps += __shfl_xor(ps, m, 32);
        pd += __shfl_xor(pd, m, 32);
    }
    if (lane == 0) { hs[gw] = ps; hd[gw] = pd; }
}

__global__ void gat_edge1(const float* __restrict__ hs, const float* __restrict__ hd,
                          const float* __restrict__ tb, const int* __restrict__ src,
                          const int* __restrict__ dst, const int* __restrict__ typ,
                          float* __restrict__ eraw, float* __restrict__ emax)
{
    long e = (long)blockIdx.x * blockDim.x + threadIdx.x;
    if (e >= EE) return;
    float x = hs[src[e]] + hd[dst[e]] + tb[typ[e]];
    x = (x > 0.0f) ? x : 0.2f * x;
    eraw[e] = x;
    atomicMaxF(&emax[dst[e]], x);
}

__global__ void gat_edge2(const float* __restrict__ eraw, const float* __restrict__ emax,
                          const int* __restrict__ dst, float* __restrict__ ex,
                          float* __restrict__ den)
{
    long e = (long)blockIdx.x * blockDim.x + threadIdx.x;
    if (e >= EE) return;
    float x = expf(eraw[e] - emax[dst[e]]);
    ex[e] = x;
    atomicAdd(&den[dst[e]], x);
}

__global__ __launch_bounds__(256)
void gat_edge3(const float* __restrict__ h, const float* __restrict__ ex,
               const float* __restrict__ den, const int* __restrict__ src,
               const int* __restrict__ dst, float* __restrict__ out)
{
    long idx = (long)blockIdx.x * blockDim.x + threadIdx.x;
    int e = (int)(idx >> 5), q = (int)(idx & 31);
    if (e >= EE) return;
    float al = ex[e] / fmaxf(den[dst[e]], 1e-9f);
    float4 v = ((const float4*)(h + (long)src[e] * HD))[q];
    float* o = out + (long)dst[e] * HD + q * 4;
    atomicAdd(o + 0, v.x * al); atomicAdd(o + 1, v.y * al);
    atomicAdd(o + 2, v.z * al); atomicAdd(o + 3, v.w * al);
}

// ---------------------------------------------------------------------------
// Final head
// ---------------------------------------------------------------------------
__global__ void concat2(const float* __restrict__ gz, const float* __restrict__ gj,
                        float* __restrict__ out)
{
    long i = (long)blockIdx.x * blockDim.x + threadIdx.x;
    if (i >= (long)NN * 256) return;
    long n = i >> 8; int c = (int)(i & 255);
    out[i] = (c < 128) ? gz[n * HD + c] : gj[n * HD + (c - 128)];
}

__global__ void head_kernel(const float* __restrict__ x, const float* __restrict__ W,
                            const float* __restrict__ b, float* __restrict__ out)
{
    int n = blockIdx.x * blockDim.x + threadIdx.x;
    if (n >= NN) return;
    float z[CC];
#pragma unroll
    for (int c = 0; c < CC; ++c) z[c] = b[c];
    const float* xr = x + (long)n * HD;
    for (int k = 0; k < HD; ++k) {
        float xv = xr[k];
#pragma unroll
        for (int c = 0; c < CC; ++c) z[c] += xv * W[k * CC + c];
    }
    float mx = z[0];
#pragma unroll
    for (int c = 1; c < CC; ++c) mx = fmaxf(mx, z[c]);
    float s = 0.0f;
#pragma unroll
    for (int c = 0; c < CC; ++c) s += expf(z[c] - mx);
    float lse = mx + logf(s);
#pragma unroll
    for (int c = 0; c < CC; ++c) out[(long)n * CC + c] = z[c] - lse;
}

// ---------------------------------------------------------------------------
// Launcher
// ---------------------------------------------------------------------------
extern "C" void kernel_launch(void* const* d_in, const int* in_sizes, int n_in,
                              void* d_out, int out_size, void* d_ws, size_t ws_size,
                              hipStream_t stream)
{
    (void)in_sizes; (void)n_in; (void)out_size; (void)ws_size;
    #define IN(i) ((const float*)d_in[(i)])
    const float* out_features = IN(0);
    const float* feat[3]  = { IN(1), IN(2), IN(3) };
    const int    featK[3] = { 100, 768, 512 };
    const float* encW[3]  = { IN(4), IN(6), IN(8) };
    const float* encB[3]  = { IN(5), IN(7), IN(9) };
    const float *p1W[4], *p1B[4], *p2W[4], *p2B[4], *fW[4], *fB[4];
    for (int m = 0; m < 4; ++m) {
        p1W[m] = IN(10 + 6 * m); p1B[m] = IN(11 + 6 * m);
        p2W[m] = IN(12 + 6 * m); p2B[m] = IN(13 + 6 * m);
        fW[m]  = IN(14 + 6 * m); fB[m]  = IN(15 + 6 * m);
    }
    const float* waW = IN(34); const float* wab = IN(35);
    const float* wlW = IN(36); const float* wlb = IN(37);
    const float* wvW = IN(38); const float* wvb = IN(39);
    const float* rgcn_W1 = IN(40); const float* rgcn_S1 = IN(41);
    const float* rgcn_W2 = IN(42); const float* rgcn_S2 = IN(43);
    const float* gat_W1 = IN(44); const float* gat_as1 = IN(45);
    const float* gat_ad1 = IN(46); const float* gat_tb1 = IN(47);
    const float* gat_W2 = IN(48); const float* gat_as2 = IN(49);
    const float* gat_ad2 = IN(50); const float* gat_tb2 = IN(51);
    const float* po1_W = IN(52); const float* po1_b = IN(53);
    const float* po2_W = IN(54); const float* po2_b = IN(55);
    const float* outW = IN(56); const float* outB = IN(57);
    const int* e_src = (const int*)d_in[58];
    const int* e_dst = (const int*)d_in[59];
    const int* e_typ = (const int*)d_in[60];

    // workspace carve-out (~348 MB)
    char* wsB = (char*)d_ws;
    size_t off = 0;
    auto carveF = [&](long nf) { float* p = (float*)(wsB + off); off += (size_t)nf * 4; return p; };
    auto carveH = [&](long nh) { _Float16* p = (_Float16*)(wsB + off); off += (size_t)nh * 2;
                                 off = (off + 15) & ~(size_t)15; return p; };
    const long MH = (long)MM * HD, NH = (long)NN * HD;
    float* P0 = carveF(MH);   // nodes
    float* P1 = carveF(MH);   // tmp / h   (lo=j_con, hi=j_a early)
    float* P2 = carveF(MH);   // agg / g_j (lo=j_l, hi=j_v early)
    float* P3 = carveF(MH);   // self / g_z (lo=dw_c, hi=dw_a early)
    float* P4 = carveF(MH);   // x2 / concat (lo=dw_l, hi=dw_v early)
    float* deg  = carveF(MM);
    float* hs   = carveF(MM);
    float* hd   = carveF(MM);
    float* emax = carveF(MM);
    float* den  = carveF(MM);
    float* eraw = carveF(EE);
    float* exb  = carveF(EE);

    const int T = 256;
    auto eb = [&](long n) { return (int)((n + T - 1) / T); };
    // convert each weight W[K][128] -> transposed, K-padded f16
    auto cvt = [&](const float* W, int K) {
        int Kp = (K + 31) & ~31;
        _Float16* d = carveH((long)Kp * 128);
        cvt_wt<<<eb((long)Kp * 128), T, 0, stream>>>(W, d, K, Kp);
        return d;
    };
    _Float16 *wtEnc[3], *wtP1[4], *wtP2[4], *wtF[4], *wtRG1[4], *wtRG2[4];
    for (int m = 0; m < 3; ++m) wtEnc[m] = cvt(encW[m], featK[m]);
    for (int m = 0; m < 4; ++m) { wtP1[m] = cvt(p1W[m], HD); wtP2[m] = cvt(p2W[m], HD); wtF[m] = cvt(fW[m], HD); }
    for (int r = 0; r < 4; ++r) { wtRG1[r] = cvt(rgcn_W1 + (long)r * HD * HD, HD);
                                  wtRG2[r] = cvt(rgcn_W2 + (long)r * HD * HD, HD); }
    _Float16* wtRS1 = cvt(rgcn_S1, HD);
    _Float16* wtRS2 = cvt(rgcn_S2, HD);
    _Float16* wtG1  = cvt(gat_W1, HD);
    _Float16* wtG2  = cvt(gat_W2, HD);
    _Float16* wtPo1 = cvt(po1_W, 256);
    _Float16* wtPo2 = cvt(po2_W, HD);

    float* encb = P0;            // N x H scratch
    float* hidb = P0 + NH;       // N x H scratch
    float* jb[4]  = { P1, P1 + NH, P2, P2 + NH };   // j_con, j_a, j_l, j_v
    float* dwb[4] = { P3, P3 + NH, P4, P4 + NH };   // dw_c, dw_a, dw_l, dw_v

    const int gN = NN / 128, gM = MM / 128;

    // ---- encoders + judge projections ----
    for (int m = 0; m < 3; ++m) {
        gemm_h128<<<gN, T, 0, stream>>>(feat[m], featK[m], featK[m], wtEnc[m], encB[m], encb, 1);
        gemm_h128<<<gN, T, 0, stream>>>(encb, HD, HD, wtP1[m + 1], p1B[m + 1], hidb, 1);
        gemm_h128<<<gN, T, 0, stream>>>(hidb, HD, HD, wtP2[m + 1], p2B[m + 1], jb[m + 1], 0);
    }
    gemm_h128<<<gN, T, 0, stream>>>(out_features, HD, HD, wtP1[0], p1B[0], hidb, 1);
    gemm_h128<<<gN, T, 0, stream>>>(hidb, HD, HD, wtP2[0], p2B[0], jb[0], 0);

    // ---- gating fusion ----
    for (int m = 0; m < 4; ++m)
        gemm_h128<<<gN, T, 0, stream>>>(jb[m], HD, HD, wtF[m], fB[m], dwb[m], 0);
    gating_kernel<<<(NN * 32) / T, T, 0, stream>>>(jb[0], jb[1], jb[2], jb[3],
        dwb[0], dwb[1], dwb[2], dwb[3], waW, wab, wlW, wlb, wvW, wvb, P0 + NH);
    copy_f32<<<eb(NH), T, 0, stream>>>(out_features, P0, NH);

    // ---- degree (shared by both RGCN layers) ----
    fill_f32<<<eb(MM), T, 0, stream>>>(deg, 0.0f, MM);
    deg_kernel<<<eb(EE), T, 0, stream>>>(e_dst, deg);

    // ---- RGCN layer 1 -> relu -> P4 ----
    fill_f32<<<eb(MH), T, 0, stream>>>(P2, 0.0f, MH);
    for (int r = 0; r < 4; ++r) {
        gemm_h128<<<gM, T, 0, stream>>>(P0, HD, HD, wtRG1[r], nullptr, P1, 0);
        rgcn_scatter<<<(EE * 32) / T, T, 0, stream>>>(P1, P2, e_src, e_dst, e_typ, r);
    }
    gemm_h128<<<gM, T, 0, stream>>>(P0, HD, HD, wtRS1, nullptr, P3, 0);
    rgcn_combine<<<eb(MH), T, 0, stream>>>(P2, P3, deg, P4, 1, MH);

    // ---- RGCN layer 2 -> g_j in P2[:N] ----
    fill_f32<<<eb(MH), T, 0, stream>>>(P2, 0.0f, MH);
    for (int r = 0; r < 4; ++r) {
        gemm_h128<<<gM, T, 0, stream>>>(P4, HD, HD, wtRG2[r], nullptr, P1, 0);
        rgcn_scatter<<<(EE * 32) / T, T, 0, stream>>>(P1, P2, e_src, e_dst, e_typ, r);
    }
    gemm_h128<<<gM, T, 0, stream>>>(P4, HD, HD, wtRS2, nullptr, P3, 0);
    rgcn_combine<<<eb(MH), T, 0, stream>>>(P2, P3, deg, P2, 0, MH);

    // ---- GAT layer 1 -> relu -> P4 ----
    gemm_h128<<<gM, T, 0, stream>>>(P0, HD, HD, wtG1, nullptr, P1, 0);
    gat_scores<<<(MM * 32) / T, T, 0, stream>>>(P1, gat_as1, gat_ad1, hs, hd, MM);
    fill_f32<<<eb(MM), T, 0, stream>>>(emax, -3.0e38f, MM);
    fill_f32<<<eb(MM), T, 0, stream>>>(den, 0.0f, MM);
    gat_edge1<<<eb(EE), T, 0, stream>>>(hs, hd, gat_tb1, e_src, e_dst, e_typ, eraw, emax);
    gat_edge2<<<eb(EE), T, 0, stream>>>(eraw, emax, e_dst, exb, den);
    fill_f32<<<eb(MH), T, 0, stream>>>(P4, 0.0f, MH);
    gat_edge3<<<(EE * 32) / T, T, 0, stream>>>(P1, exb, den, e_src, e_dst, P4);
    relu_f32<<<eb(MH), T, 0, stream>>>(P4, MH);

    // ---- GAT layer 2 -> g_z in P3[:N] ----
    gemm_h128<<<gM, T, 0, stream>>>(P4, HD, HD, wtG2, nullptr, P1, 0);
    gat_scores<<<(MM * 32) / T, T, 0, stream>>>(P1, gat_as2, gat_ad2, hs, hd, MM);
    fill_f32<<<eb(MM), T, 0, stream>>>(emax, -3.0e38f, MM);
    fill_f32<<<eb(MM), T, 0, stream>>>(den, 0.0f, MM);
    gat_edge1<<<eb(EE), T, 0, stream>>>(hs, hd, gat_tb2, e_src, e_dst, e_typ, eraw, emax);
    gat_edge2<<<eb(EE), T, 0, stream>>>(eraw, emax, e_dst, exb, den);
    fill_f32<<<eb(MH), T, 0, stream>>>(P3, 0.0f, MH);
    gat_edge3<<<(EE * 32) / T, T, 0, stream>>>(P1, exb, den, e_src, e_dst, P3);

    // ---- final head ----
    concat2<<<eb((long)NN * 256), T, 0, stream>>>(P3, P2, P4);
    gemm_h128<<<gN, T, 0, stream>>>(P4, 256, 256, wtPo1, po1_b, P1, 1);
    gemm_h128<<<gN, T, 0, stream>>>(P1, HD, HD, wtPo2, po2_b, P0, 0);
    head_kernel<<<eb(NN), T, 0, stream>>>(P0, outW, outB, (float*)d_out);
    #undef IN
}
